// AIGGenerator_55482387530047
// MI455X (gfx1250) — matrix operand order repaired and checked by reading, not compiled, not run on hardware
//
#include <hip/hip_runtime.h>
#include <hip/hip_bf16.h>

// Problem constants (reference: N=12288, E=196608, H=Z=128) — H hardcoded,
// N/E taken from in_sizes.
#define HD 128
#define NEGV (-1.0e9f)

typedef __attribute__((ext_vector_type(16))) _Float16 v16h;
typedef __attribute__((ext_vector_type(8)))  _Float16 v8h;
typedef __attribute__((ext_vector_type(8)))  float    v8f;

// Build the A-operand v16h for one 16x32 f16 tile:
// lane (m, ks): element s holds K = base+s (contiguous, same order as B)
// where base = kstep*32 + 16*ks -> two adjacent 16B chunks.
__device__ __forceinline__ v16h load_a16(const _Float16* __restrict__ rowp,
                                         int kstep, int ks) {
  const _Float16* p = rowp + kstep * 32 + 16 * ks;
  v8h lo = *(const v8h*)(p);
  v8h hi = *(const v8h*)(p + 8);
  return __builtin_shufflevector(lo, hi, 0, 1, 2, 3, 4, 5, 6, 7,
                                 8, 9, 10, 11, 12, 13, 14, 15);
}

// B operand: lane (n, ks): VGPR j holds K = kstep*32 + 16*ks + 2j,+1 ->
// 16 contiguous halves of the N-major (transposed) matrix row n.
__device__ __forceinline__ v16h load_b16(const _Float16* __restrict__ rowp,
                                         int kstep, int ks) {
  return *(const v16h*)(rowp + kstep * 32 + 16 * ks);
}

// ---------------------------------------------------------------------------
// Phase 0: init deg=1 (self loop), zero agg buffers
// ---------------------------------------------------------------------------
__global__ void k_init(float* __restrict__ deg, float* __restrict__ agg1,
                       float* __restrict__ agg2, int n) {
  int total = n * HD;
  for (int i = blockIdx.x * blockDim.x + threadIdx.x; i < total;
       i += gridDim.x * blockDim.x) {
    agg2[i] = 0.f;
    if (i < n) deg[i] = 1.f;
    if (i < 2 * n) agg1[i] = 0.f;
  }
}

__global__ void k_deg(const int* __restrict__ dst, float* __restrict__ deg, int e) {
  int i = blockIdx.x * blockDim.x + threadIdx.x;
  if (i < e) atomicAdd(&deg[dst[i]], 1.f);
}

__global__ void k_isq(const float* __restrict__ deg, float* __restrict__ isq, int n) {
  int i = blockIdx.x * blockDim.x + threadIdx.x;
  if (i < n) isq[i] = rsqrtf(deg[i]);   // deg >= 1 always
}

// ---------------------------------------------------------------------------
// GCN layer 1: scatter x[s]*w into agg1, then tiny [N,2]@[2,128] transform
// ---------------------------------------------------------------------------
__global__ void k_scatter1(const int* __restrict__ src, const int* __restrict__ dst,
                           const float* __restrict__ isq, const float* __restrict__ x,
                           float* __restrict__ agg1, int e) {
  int i = blockIdx.x * blockDim.x + threadIdx.x;
  if (i >= e) return;
  int s = src[i], d = dst[i];
  float w = isq[s] * isq[d];
  atomicAdd(&agg1[d * 2 + 0], x[s * 2 + 0] * w);
  atomicAdd(&agg1[d * 2 + 1], x[s * 2 + 1] * w);
}

__global__ void __launch_bounds__(HD)
k_h1(const float* __restrict__ agg1, const float* __restrict__ x,
     const float* __restrict__ isq, const float* __restrict__ W1,
     const float* __restrict__ b1, float* __restrict__ h1) {
  int n = blockIdx.x, o = threadIdx.x;
  float wl = isq[n] * isq[n];                     // self-loop weight
  float a0 = agg1[n * 2 + 0] + x[n * 2 + 0] * wl;
  float a1 = agg1[n * 2 + 1] + x[n * 2 + 1] * wl;
  float v = a0 * W1[0 * HD + o] + a1 * W1[1 * HD + o] + b1[o];
  h1[(size_t)n * HD + o] = v > 0.f ? v : 0.f;
}

// ---------------------------------------------------------------------------
// GCN layer 2 scatter: agg2[d] += h1[s]*w (4 features per thread)
// ---------------------------------------------------------------------------
__global__ void k_scatter2(const int* __restrict__ src, const int* __restrict__ dst,
                           const float* __restrict__ isq, const float* __restrict__ h1,
                           float* __restrict__ agg2, int e) {
  int t = blockIdx.x * blockDim.x + threadIdx.x;
  if (t >= e * 32) return;
  int ei = t >> 5;
  int f  = (t & 31) * 4;
  int s = src[ei], d = dst[ei];
  float w = isq[s] * isq[d];
  const float4 hs = *(const float4*)(h1 + (size_t)s * HD + f);
  float* ad = agg2 + (size_t)d * HD + f;
  atomicAdd(&ad[0], hs.x * w);
  atomicAdd(&ad[1], hs.y * w);
  atomicAdd(&ad[2], hs.z * w);
  atomicAdd(&ad[3], hs.w * w);
}

// add self loop, convert to f16 for WMMA
__global__ void k_selfcvt(const float* __restrict__ h1, const float* __restrict__ isq,
                          const float* __restrict__ agg2, _Float16* __restrict__ agg2h,
                          int n) {
  int t = blockIdx.x * blockDim.x + threadIdx.x;
  if (t >= n * HD) return;
  int node = t >> 7;
  float v = agg2[t] + h1[t] * isq[node] * isq[node];
  agg2h[t] = (_Float16)v;
}

// Convert the five 128x128 weight blocks to f16 *transposed* (N-major) so the
// WMMA B operand is 16 contiguous halves per lane (-> global_load_b128 pairs).
// Wm1: top 128 rows only (z part folded into cm).
__global__ void k_wcvt(const float* __restrict__ W2, const float* __restrict__ Wm1,
                       const float* __restrict__ Wm2, const float* __restrict__ Ws,
                       const float* __restrict__ Wt, _Float16* __restrict__ w16) {
  int i = blockIdx.x * blockDim.x + threadIdx.x;
  if (i >= HD * HD) return;
  int k = i >> 7, nn = i & 127;
  int ti = nn * HD + k;            // transposed position
  w16[0 * HD * HD + ti] = (_Float16)W2[i];
  w16[1 * HD * HD + ti] = (_Float16)Wm1[i];
  w16[2 * HD * HD + ti] = (_Float16)Wm2[i];
  w16[3 * HD * HD + ti] = (_Float16)Ws[i];
  w16[4 * HD * HD + ti] = (_Float16)Wt[i];
}

// cm = z @ Wm1[128:,:] + bm1 ; ci = z @ Wi1[256:,:] + bi1  (z broadcast trick)
__global__ void __launch_bounds__(HD)
k_cvec(const float* __restrict__ z, const float* __restrict__ Wm1,
       const float* __restrict__ bm1, const float* __restrict__ Wi1,
       const float* __restrict__ bi1, float* __restrict__ cm, float* __restrict__ ci) {
  int o = threadIdx.x;
  if (blockIdx.x == 0) {
    float acc = bm1[o];
    for (int j = 0; j < HD; ++j) acc += z[j] * Wm1[(HD + j) * HD + o];
    cm[o] = acc;
  } else {
    float acc = bi1[o];
    for (int j = 0; j < HD; ++j) acc += z[j] * Wi1[(2 * HD + j) * HD + o];
    ci[o] = acc;
  }
}

// ---------------------------------------------------------------------------
// WMMA GEMM: out = [relu](A[M,128] @ W + bias), W given f16-transposed [N,K].
// One 16-row tile per block (4 waves x two 16-col tiles; K=128 -> 4 WMMAs).
// ---------------------------------------------------------------------------
__global__ void __launch_bounds__(128)
k_gemm128(const _Float16* __restrict__ A, const _Float16* __restrict__ Bt,
          const float* __restrict__ bias, float* __restrict__ outF,
          _Float16* __restrict__ outH, int relu) {
  const int row0 = blockIdx.x * 16;
  const int wave = threadIdx.x >> 5;
  const int lane = threadIdx.x & 31;
  const int m  = lane & 15;       // row (A) / col (B,C)
  const int ks = lane >> 4;       // half-wave select

  const _Float16* arow = A + (size_t)(row0 + m) * HD;
  v16h a[4];
#pragma unroll
  for (int kstep = 0; kstep < 4; ++kstep) a[kstep] = load_a16(arow, kstep, ks);

#pragma unroll
  for (int ct = 0; ct < 2; ++ct) {
    const int col0 = wave * 32 + ct * 16;
    const _Float16* brow = Bt + (size_t)(col0 + m) * HD;
    v8f c = {};
#pragma unroll
    for (int kstep = 0; kstep < 4; ++kstep) {
      v16h b = load_b16(brow, kstep, ks);
      c = __builtin_amdgcn_wmma_f32_16x16x32_f16(false, a[kstep], false, b,
                                                 (short)0, c, false, false);
    }
    const int col = col0 + m;
#pragma unroll
    for (int j = 0; j < 8; ++j) {
      int rowm = j + 8 * ks;
      float v = c[j] + (bias ? bias[col] : 0.f);
      if (relu && v < 0.f) v = 0.f;
      if (outF) outF[(size_t)(row0 + rowm) * HD + col] = v;
      if (outH) outH[(size_t)(row0 + rowm) * HD + col] = (_Float16)v;
    }
  }
}

// ---------------------------------------------------------------------------
// Fused St = T @ S^T with depth mask + per-row top-2 (never materializes NxN).
// 32 target rows per block: each wave computes TWO 16x16 row-tiles against one
// shared 16-col B tile -> 8 WMMAs per 8 global_load_b128. S (f16, 3MB) stays
// L2-resident, so the loop is WMMA-throughput-bound.
// ---------------------------------------------------------------------------
__global__ void __launch_bounds__(128)
k_topk(const _Float16* __restrict__ T, const _Float16* __restrict__ S,
       const int* __restrict__ depth, float* __restrict__ valsO,
       int* __restrict__ idxO, int n) {
  __shared__ float sv[32][128];
  __shared__ int   si[32][128];
  __shared__ int   dv[32];

  const int row0 = blockIdx.x * 32;
  const int wave = threadIdx.x >> 5;
  const int lane = threadIdx.x & 31;
  const int m  = lane & 15;
  const int ks = lane >> 4;

  if (threadIdx.x < 32) dv[threadIdx.x] = depth[row0 + threadIdx.x];
  __syncthreads();

  v16h a[2][4];
#pragma unroll
  for (int t = 0; t < 2; ++t) {
    const _Float16* arow = T + (size_t)(row0 + t * 16 + m) * HD;
#pragma unroll
    for (int kstep = 0; kstep < 4; ++kstep) a[t][kstep] = load_a16(arow, kstep, ks);
  }

  float v1[2][8], v2[2][8];
  int   i1[2][8], i2[2][8];
#pragma unroll
  for (int t = 0; t < 2; ++t)
#pragma unroll
    for (int j = 0; j < 8; ++j) {
      v1[t][j] = v2[t][j] = -3.0e38f;
      i1[t][j] = i2[t][j] = 0x7fffffff;
    }

  for (int col0 = wave * 16; col0 < n; col0 += 64) {
    if (col0 + 64 < n)
      __builtin_prefetch(&S[(size_t)(col0 + 64 + m) * HD], 0, 0);  // global_prefetch_b8
    const _Float16* brow = S + (size_t)(col0 + m) * HD;
    v8f c0 = {}, c1 = {};
#pragma unroll
    for (int kstep = 0; kstep < 4; ++kstep) {
      v16h b = load_b16(brow, kstep, ks);
      c0 = __builtin_amdgcn_wmma_f32_16x16x32_f16(false, a[0][kstep], false, b,
                                                  (short)0, c0, false, false);
      c1 = __builtin_amdgcn_wmma_f32_16x16x32_f16(false, a[1][kstep], false, b,
                                                  (short)0, c1, false, false);
    }
    const int u  = col0 + m;
    const int du = depth[u];
#pragma unroll
    for (int t = 0; t < 2; ++t) {
#pragma unroll
      for (int j = 0; j < 8; ++j) {
        int rowm = t * 16 + j + 8 * ks;
        float cv = (t == 0) ? c0[j] : c1[j];
        float val = (du < dv[rowm]) ? cv : NEGV;
        // per-lane columns strictly increase -> '>' keeps lowest index on ties
        if (val > v1[t][j]) {
          v2[t][j] = v1[t][j]; i2[t][j] = i1[t][j];
          v1[t][j] = val;      i1[t][j] = u;
        } else if (val > v2[t][j]) {
          v2[t][j] = val; i2[t][j] = u;
        }
      }
    }
  }

#pragma unroll
  for (int t = 0; t < 2; ++t)
#pragma unroll
    for (int j = 0; j < 8; ++j) {
      int rowm = t * 16 + j + 8 * ks;
      int slot = wave * 16 + m;
      sv[rowm][slot * 2]     = v1[t][j]; si[rowm][slot * 2]     = i1[t][j];
      sv[rowm][slot * 2 + 1] = v2[t][j]; si[rowm][slot * 2 + 1] = i2[t][j];
    }
  __syncthreads();

  if (threadIdx.x < 32) {
    int rm = threadIdx.x;
    float b1v = -3.0e38f, b2v = -3.0e38f;
    int   b1i = 0x7fffffff, b2i = 0x7fffffff;
    for (int q = 0; q < 128; ++q) {
      float v = sv[rm][q]; int ii = si[rm][q];
      if (v > b1v || (v == b1v && ii < b1i)) { b2v = b1v; b2i = b1i; b1v = v; b1i = ii; }
      else if (v > b2v || (v == b2v && ii < b2i)) { b2v = v; b2i = ii; }
    }
    valsO[(row0 + rm) * 2 + 0] = b1v;
    valsO[(row0 + rm) * 2 + 1] = b2v;
    idxO[(row0 + rm) * 2 + 0]  = b1i;
    idxO[(row0 + rm) * 2 + 1]  = b2i;
  }
}

// ---------------------------------------------------------------------------
// Inversion-bit MLP + edge_valid + output packing.
// One block per (target v, slot k). d_out = [inv_prob | vals | idx | edge_valid].
// ---------------------------------------------------------------------------
__global__ void __launch_bounds__(HD)
k_final(const float* __restrict__ hf, const float* __restrict__ vals,
        const int* __restrict__ idxb, const int* __restrict__ ntype,
        const int* __restrict__ ndepth, const float* __restrict__ Wi1,
        const float* __restrict__ ci, const float* __restrict__ Wi2,
        const float* __restrict__ bi2, float* __restrict__ out, int n) {
  __shared__ float red[HD];
  const int p = blockIdx.x;            // 0 .. 2n-1
  const int v = p >> 1, k = p & 1;
  const int o = threadIdx.x;
  const int u = idxb[p];
  const float* hu = hf + (size_t)u * HD;
  const float* hv = hf + (size_t)v * HD;
  float acc = ci[o];
  for (int j = 0; j < HD; ++j) acc += hu[j] * Wi1[j * HD + o];
  for (int j = 0; j < HD; ++j) acc += hv[j] * Wi1[(HD + j) * HD + o];
  acc = acc > 0.f ? acc : 0.f;
  red[o] = acc * Wi2[o];
  __syncthreads();
  for (int st = HD / 2; st > 0; st >>= 1) {
    if (o < st) red[o] += red[o + st];
    __syncthreads();
  }
  if (o == 0) {
    float logit = red[0] + bi2[0];
    float pr = 1.f / (1.f + __expf(-logit));
    int t = ntype[v], d = ndepth[v];
    bool tv = (t != 0) && (d > 0) && (vals[v * 2 + 0] > NEGV * 0.5f);
    bool ev = (k == 0) ? tv : (tv && (t == 2));
    out[p]             = pr;               // inv_prob
    out[2 * n + p]     = vals[p];          // vals
    out[4 * n + p]     = (float)idxb[p];   // idx
    out[6 * n + p]     = ev ? 1.f : 0.f;   // edge_valid
  }
}

// ---------------------------------------------------------------------------
extern "C" void kernel_launch(void* const* d_in, const int* in_sizes, int n_in,
                              void* d_out, int out_size, void* d_ws, size_t ws_size,
                              hipStream_t stream) {
  const float* x    = (const float*)d_in[0];
  const float* z    = (const float*)d_in[1];
  const int*   ntyp = (const int*)  d_in[2];
  const int*   ndep = (const int*)  d_in[3];
  const int*   eidx = (const int*)  d_in[4];
  const float* W1   = (const float*)d_in[5];
  const float* b1   = (const float*)d_in[6];
  const float* W2   = (const float*)d_in[7];
  const float* b2   = (const float*)d_in[8];
  const float* Wm1  = (const float*)d_in[9];
  const float* bm1  = (const float*)d_in[10];
  const float* Wm2  = (const float*)d_in[11];
  const float* bm2  = (const float*)d_in[12];
  const float* Ws   = (const float*)d_in[13];
  const float* Wt   = (const float*)d_in[14];
  const float* Wi1  = (const float*)d_in[15];
  const float* bi1  = (const float*)d_in[16];
  const float* Wi2  = (const float*)d_in[17];
  const float* bi2  = (const float*)d_in[18];

  const int n = in_sizes[0] / 2;    // x is [N,2]
  const int e = in_sizes[4] / 2;    // edge_index is [2,E]
  const int* src = eidx;
  const int* dst = eidx + e;

  char* ws = (char*)d_ws;
  size_t off = 0;
  auto alloc = [&](size_t bytes) -> void* {
    void* p = ws + off;
    off = (off + bytes + 255) & ~(size_t)255;
    return p;
  };
  float*    deg   = (float*)   alloc((size_t)n * 4);
  float*    isq   = (float*)   alloc((size_t)n * 4);
  float*    agg1  = (float*)   alloc((size_t)n * 2 * 4);
  float*    h1    = (float*)   alloc((size_t)n * HD * 4);
  float*    agg2  = (float*)   alloc((size_t)n * HD * 4);
  _Float16* agg2h = (_Float16*)alloc((size_t)n * HD * 2);
  _Float16* h2h   = (_Float16*)alloc((size_t)n * HD * 2);
  _Float16* m1h   = (_Float16*)alloc((size_t)n * HD * 2);
  float*    hf    = (float*)   alloc((size_t)n * HD * 4);
  _Float16* hh    = (_Float16*)alloc((size_t)n * HD * 2);
  _Float16* th    = (_Float16*)alloc((size_t)n * HD * 2);
  _Float16* sh    = (_Float16*)alloc((size_t)n * HD * 2);
  _Float16* w16   = (_Float16*)alloc((size_t)5 * HD * HD * 2);
  float*    cm    = (float*)   alloc(HD * 4);
  float*    civ   = (float*)   alloc(HD * 4);
  float*    valsb = (float*)   alloc((size_t)n * 2 * 4);
  int*      idxb  = (int*)     alloc((size_t)n * 2 * 4);
  (void)ws_size; (void)n_in; (void)out_size;

  _Float16* W2h  = w16 + 0 * HD * HD;   // all stored transposed [N][K]
  _Float16* Wm1h = w16 + 1 * HD * HD;
  _Float16* Wm2h = w16 + 2 * HD * HD;
  _Float16* Wsh  = w16 + 3 * HD * HD;
  _Float16* Wth  = w16 + 4 * HD * HD;

  // Phase 0: init + degrees
  k_init<<<2048, 256, 0, stream>>>(deg, agg1, agg2, n);
  k_deg<<<(e + 255) / 256, 256, 0, stream>>>(dst, deg, e);
  k_isq<<<(n + 255) / 256, 256, 0, stream>>>(deg, isq, n);

  // weight prep (independent of graph work)
  k_wcvt<<<(HD * HD + 255) / 256, 256, 0, stream>>>(W2, Wm1, Wm2, Ws, Wt, w16);
  k_cvec<<<2, HD, 0, stream>>>(z, Wm1, bm1, Wi1, bi1, cm, civ);

  // GCN layer 1
  k_scatter1<<<(e + 255) / 256, 256, 0, stream>>>(src, dst, isq, x, agg1, e);
  k_h1<<<n, HD, 0, stream>>>(agg1, x, isq, W1, b1, h1);

  // GCN layer 2 scatter + transform (WMMA)
  k_scatter2<<<((size_t)e * 32 + 255) / 256, 256, 0, stream>>>(src, dst, isq, h1, agg2, e);
  k_selfcvt<<<((size_t)n * HD + 255) / 256, 256, 0, stream>>>(h1, isq, agg2, agg2h, n);
  k_gemm128<<<n / 16, 128, 0, stream>>>(agg2h, W2h, b2, nullptr, h2h, 1);      // h2 = relu
  // node MLP (z folded into cm)
  k_gemm128<<<n / 16, 128, 0, stream>>>(h2h, Wm1h, cm, nullptr, m1h, 1);       // relu
  k_gemm128<<<n / 16, 128, 0, stream>>>(m1h, Wm2h, bm2, hf, hh, 0);            // h (f32+f16)
  // score projections
  k_gemm128<<<n / 16, 128, 0, stream>>>(hh, Wth, nullptr, nullptr, th, 0);     // t = h@Wt
  k_gemm128<<<n / 16, 128, 0, stream>>>(hh, Wsh, nullptr, nullptr, sh, 0);     // s = h@Ws

  // fused masked score GEMM + top-2 (38.7 GFLOP, S f16 L2-resident)
  k_topk<<<n / 32, 128, 0, stream>>>(th, sh, ndep, valsb, idxb, n);

  // inversion MLP + edge_valid + pack outputs
  k_final<<<2 * n, HD, 0, stream>>>(hf, valsb, idxb, ntyp, ndep, Wi1, civ, Wi2,
                                    bi2, (float*)d_out, n);
}
